// CODEXDose_32143535243760
// MI455X (gfx1250) — matrix operand order repaired and checked by reading, not compiled, not run on hardware
//
#include <hip/hip_runtime.h>
#include <hip/hip_bf16.h>
#include <math.h>

typedef __attribute__((ext_vector_type(2)))  __bf16 v2bf;
typedef __attribute__((ext_vector_type(8)))  __bf16 v8bf;
typedef __attribute__((ext_vector_type(16))) __bf16 v16bf;
typedef __attribute__((ext_vector_type(8)))  float  v8f;

#define BM 128
#define BN 64
#define BK 32
#define LDA_S 40               // padded row stride (bf16) = 80B, keeps 16B alignment
#define LDB_S 40
#define A_TILE (BM * LDA_S)
#define B_TILE (BN * LDB_S)

// ACT: 0 = relu, 1 = none, 2 = final (col<HALF ? x : softplus(x)+1e-3)
// ABF16:  A matrix is bf16 (intermediate activations); else f32
// OUTBF16: write C as bf16 (intermediates); else f32
template<int ACT, bool ABF16, bool OUTBF16>
__global__ __launch_bounds__(256)
void gemm_wmma(const void* __restrict__ Av, const float* __restrict__ W,
               const float* __restrict__ bias, void* __restrict__ Cv,
               int K, int N, int HALF)
{
    __shared__ alignas(16) __bf16 As[2 * A_TILE];   // double-buffered
    __shared__ alignas(16) __bf16 Bs[2 * B_TILE];   // transposed: Bs[n][k]

    const float*  Af = (const float*)Av;
    const __bf16* Ab = (const __bf16*)Av;
    float*  Cf = (float*)Cv;
    __bf16* Cb = (__bf16*)Cv;

    const int tid  = threadIdx.x;
    const int lane = tid & 31;
    const int wave = tid >> 5;
    const int wm   = wave & 3;           // 4 waves along M
    const int wn   = wave >> 2;          // 2 waves along N
    const int block_m = blockIdx.y * BM;
    const int block_n = blockIdx.x * BN;
    const bool nfull = (block_n + BN) <= N;

    v8f acc[2][2];
    #pragma unroll
    for (int i = 0; i < 2; ++i)
        #pragma unroll
        for (int j = 0; j < 2; ++j)
            #pragma unroll
            for (int v = 0; v < 8; ++v) acc[i][j][v] = 0.0f;

    // ISA 7.12.2 lane mappings (wave32)
    const int a_m  = lane & 15;
    const int a_kb = (lane < 16) ? 0 : 8;
    const int b_n  = lane & 15;
    const int b_kb = (lane < 16) ? 0 : 8;
    const int c_mo = (lane < 16) ? 0 : 8;
    const int c_n  = lane & 15;

    // staging coordinates
    const int ar   = tid >> 1;           // A row 0..127
    const int ac0  = (tid & 1) * 16;     // A col chunk {0,16}
    const int bc4  = (tid & 15) * 4;     // B column group base 0..60
    const int brp  = tid >> 4;           // B row-pair 0..15 (rows 2*brp, 2*brp+1)

    struct AR { float4 q0, q1, q2, q3; v8bf lo, hi; };  // raw f32 or bf16 copy
    struct BR { float4 r0, r1; };                       // raw f32, 2 rows x 4 cols

    auto loadA = [&](int k0) {
        AR r;
        if (ABF16) {
            const __bf16* src = Ab + (size_t)(block_m + ar) * K + k0 + ac0;
            r.lo = ((const v8bf*)src)[0];
            r.hi = ((const v8bf*)src)[1];
        } else {
            const float* src = Af + (size_t)(block_m + ar) * K + k0 + ac0;
            r.q0 = ((const float4*)src)[0];
            r.q1 = ((const float4*)src)[1];
            r.q2 = ((const float4*)src)[2];
            r.q3 = ((const float4*)src)[3];
        }
        return r;
    };

    auto loadB = [&](int k0) {           // full-K tiles only; N may be ragged
        BR r;
        const float* w0p = W + (size_t)(k0 + 2 * brp) * N + block_n + bc4;
        const float* w1p = w0p + N;
        if (nfull) {
            r.r0 = *(const float4*)w0p;
            r.r1 = *(const float4*)w1p;
        } else {
            float t0[4], t1[4];
            #pragma unroll
            for (int j = 0; j < 4; ++j) {
                const bool cok = (block_n + bc4 + j) < N;
                t0[j] = cok ? w0p[j] : 0.0f;
                t1[j] = cok ? w1p[j] : 0.0f;
            }
            r.r0 = make_float4(t0[0], t0[1], t0[2], t0[3]);
            r.r1 = make_float4(t1[0], t1[1], t1[2], t1[3]);
        }
        return r;
    };

    auto storeA = [&](int buf, const AR& r) {           // convert here (deferred waits)
        __bf16* dst = &As[buf * A_TILE + ar * LDA_S + ac0];
        if (ABF16) {
            *(v8bf*)dst       = r.lo;
            *(v8bf*)(dst + 8) = r.hi;
        } else {
            v8bf lo = { (__bf16)r.q0.x, (__bf16)r.q0.y, (__bf16)r.q0.z, (__bf16)r.q0.w,
                        (__bf16)r.q1.x, (__bf16)r.q1.y, (__bf16)r.q1.z, (__bf16)r.q1.w };
            v8bf hi = { (__bf16)r.q2.x, (__bf16)r.q2.y, (__bf16)r.q2.z, (__bf16)r.q2.w,
                        (__bf16)r.q3.x, (__bf16)r.q3.y, (__bf16)r.q3.z, (__bf16)r.q3.w };
            *(v8bf*)dst       = lo;
            *(v8bf*)(dst + 8) = hi;
        }
    };

    auto storeB = [&](int buf, const BR& r) {           // pack k-pairs, transposed
        __bf16* base = &Bs[buf * B_TILE + 2 * brp];
        v2bf p0 = { (__bf16)r.r0.x, (__bf16)r.r1.x };
        v2bf p1 = { (__bf16)r.r0.y, (__bf16)r.r1.y };
        v2bf p2 = { (__bf16)r.r0.z, (__bf16)r.r1.z };
        v2bf p3 = { (__bf16)r.r0.w, (__bf16)r.r1.w };
        *(v2bf*)&base[(bc4 + 0) * LDB_S] = p0;
        *(v2bf*)&base[(bc4 + 1) * LDB_S] = p1;
        *(v2bf*)&base[(bc4 + 2) * LDB_S] = p2;
        *(v2bf*)&base[(bc4 + 3) * LDB_S] = p3;
    };

    auto compute = [&](int buf) {
        v16bf afrag[2], bfrag[2];
        #pragma unroll
        for (int mi = 0; mi < 2; ++mi) {
            const __bf16* arp = &As[buf * A_TILE + (wm * 32 + mi * 16 + a_m) * LDA_S + a_kb];
            const v8bf l = *(const v8bf*)arp;        // K = kb+0..7
            const v8bf h = *(const v8bf*)(arp + 16); // K = kb+16..23
            afrag[mi] = __builtin_shufflevector(l, h, 0,1,2,3,4,5,6,7,8,9,10,11,12,13,14,15);
        }
        #pragma unroll
        for (int ni = 0; ni < 2; ++ni) {
            const __bf16* brp2 = &Bs[buf * B_TILE + (wn * 32 + ni * 16 + b_n) * LDB_S + b_kb];
            const v8bf l = *(const v8bf*)brp2;       // K = kb+0..7
            const v8bf h = *(const v8bf*)(brp2 + 16); // K = kb+16..23
            bfrag[ni] = __builtin_shufflevector(l, h, 0,1,2,3,4,5,6,7,8,9,10,11,12,13,14,15);
        }
        #pragma unroll
        for (int mi = 0; mi < 2; ++mi)
            #pragma unroll
            for (int ni = 0; ni < 2; ++ni)
                acc[mi][ni] = __builtin_amdgcn_wmma_f32_16x16x32_bf16(
                    false, afrag[mi], false, bfrag[ni],
                    (short)0, acc[mi][ni], false, false);
    };

    // guarded tail staging (K % 32 != 0; enc1 only, runs once per block)
    auto stage_tail = [&](int buf, int k0) {
        if (!ABF16) {
            const float* src = Af + (size_t)(block_m + ar) * K + k0 + ac0;
            #pragma unroll
            for (int i = 0; i < 16; ++i) {
                float v = (k0 + ac0 + i < K) ? src[i] : 0.0f;
                As[buf * A_TILE + ar * LDA_S + ac0 + i] = (__bf16)v;
            }
        }
        const float* w0p = W + (size_t)(k0 + 2 * brp) * N + block_n + bc4;
        const float* w1p = w0p + N;
        const bool k0ok = (k0 + 2 * brp)     < K;
        const bool k1ok = (k0 + 2 * brp + 1) < K;
        #pragma unroll
        for (int j = 0; j < 4; ++j) {
            const bool cok = (block_n + bc4 + j) < N;
            const float w0 = (k0ok && cok) ? w0p[j] : 0.0f;
            const float w1 = (k1ok && cok) ? w1p[j] : 0.0f;
            v2bf p = { (__bf16)w0, (__bf16)w1 };
            *(v2bf*)&Bs[buf * B_TILE + (bc4 + j) * LDB_S + 2 * brp] = p;
        }
    };

    // ---------------- software-pipelined main loop over full K tiles ----------------
    const int Kf = K & ~(BK - 1);
    const int ntiles = Kf / BK;
    int buf = 0;
    {
        AR a = loadA(0);
        BR b = loadB(0);
        storeA(0, a);
        storeB(0, b);
        __syncthreads();
        for (int t = 1; t < ntiles; ++t) {
            AR a2 = loadA(t * BK);       // issue next tile's global loads...
            BR b2 = loadB(t * BK);
            compute(buf);                // ...hidden behind current tile's WMMAs
            buf ^= 1;
            storeA(buf, a2);             // cvt + ds stores after the WMMAs
            storeB(buf, b2);
            __syncthreads();
        }
        compute(buf);
    }
    if (Kf < K) {                        // ragged K tail (once, outside hot loop)
        buf ^= 1;
        stage_tail(buf, Kf);
        __syncthreads();
        compute(buf);
    }

    // ---------------- fused epilogue ----------------
    #pragma unroll
    for (int mi = 0; mi < 2; ++mi) {
        #pragma unroll
        for (int ni = 0; ni < 2; ++ni) {
            const int col = block_n + wn * 32 + ni * 16 + c_n;
            if (col >= N) continue;
            const float bv = bias[col];
            #pragma unroll
            for (int v = 0; v < 8; ++v) {
                const int row = block_m + wm * 32 + mi * 16 + c_mo + v;
                float x = acc[mi][ni][v] + bv;
                if (ACT == 0) {
                    x = x > 0.0f ? x : 0.0f;
                } else if (ACT == 2) {
                    if (col >= HALF) {
                        x = (x > 0.0f) ? (x + log1pf(expf(-x))) : log1pf(expf(x));
                        x += 0.001f;
                    }
                }
                if (OUTBF16) Cb[(size_t)row * N + col] = (__bf16)x;
                else         Cf[(size_t)row * N + col] = x;
            }
        }
    }
}

// Per-row expert: argmax over 16 treatments, gather dose, 257x256 matvec + relu.
// Expert weights total 4.2 MB -> L2 resident; VALU is sufficient for 2.2 GFLOP.
__global__ __launch_bounds__(256)
void expert_kernel(const __bf16* __restrict__ emb,   // (B,256) bf16
                   const float* __restrict__ td,     // (B,16)
                   const float* __restrict__ ew,     // (16,257,256)
                   const float* __restrict__ eb,     // (16,256)
                   float*  __restrict__ latent_out,  // d_out tail (B,256) f32
                   __bf16* __restrict__ latent_ws)   // ws copy (B,256) bf16
{
    const int row = blockIdx.x;
    const int n   = threadIdx.x;
    __shared__ float xs[257];
    __shared__ int   t_sh;

    xs[n] = (float)emb[(size_t)row * 256 + n];
    if (n == 0) {
        const float* t = td + (size_t)row * 16;
        int best = 0; float bv = t[0];
        #pragma unroll
        for (int i = 1; i < 16; ++i) {       // first max wins (jnp.argmax)
            float v = t[i];
            if (v > bv) { bv = v; best = i; }
        }
        t_sh = best;
        xs[256] = t[best];                   // take_along_axis dose
    }
    __syncthreads();

    const int t = t_sh;
    const float* w = ew + (size_t)t * 257 * 256 + n;
    float acc = eb[t * 256 + n];
    #pragma unroll 8
    for (int i = 0; i < 257; ++i)
        acc += xs[i] * w[(size_t)i * 256];
    acc = acc > 0.0f ? acc : 0.0f;

    latent_out[(size_t)row * 256 + n] = acc;
    latent_ws [(size_t)row * 256 + n] = (__bf16)acc;
}

extern "C" void kernel_launch(void* const* d_in, const int* in_sizes, int n_in,
                              void* d_out, int out_size, void* d_ws, size_t ws_size,
                              hipStream_t stream) {
    (void)in_sizes; (void)n_in; (void)out_size; (void)ws_size;

    const int B = 16384, IN = 5000, N0 = 512, N1 = 512, N2 = 256, OUTC = 2 * IN;

    const float* input  = (const float*)d_in[0];
    const float* td     = (const float*)d_in[1];
    const float* enc_w1 = (const float*)d_in[2];
    const float* enc_b1 = (const float*)d_in[3];
    const float* enc_w2 = (const float*)d_in[4];
    const float* enc_b2 = (const float*)d_in[5];
    const float* exp_w  = (const float*)d_in[6];
    const float* exp_b  = (const float*)d_in[7];
    const float* dec_w1 = (const float*)d_in[8];
    const float* dec_b1 = (const float*)d_in[9];
    const float* dec_w2 = (const float*)d_in[10];
    const float* dec_b2 = (const float*)d_in[11];
    const float* dec_w3 = (const float*)d_in[12];
    const float* dec_b3 = (const float*)d_in[13];

    float* out        = (float*)d_out;
    float* latent_out = out + (size_t)B * OUTC;    // second tuple element

    // 32 MB of bf16 intermediates, reused -> L2 resident (192 MB L2)
    __bf16* wsb    = (__bf16*)d_ws;
    __bf16* h      = wsb;                          // (B,512)
    __bf16* emb    = wsb + (size_t)B * 512;        // (B,256)
    __bf16* latent = wsb + (size_t)B * 768;        // (B,256)
    __bf16* d1     = h;                            // reuse h slot (B,512)
    __bf16* d2     = emb;                          // reuse emb+latent (B,512)

    dim3 blk(256);

    // enc1: f32 input @ enc_w1 -> bf16 h, relu
    gemm_wmma<0,false,true><<<dim3(N0 / BN, B / BM), blk, 0, stream>>>(
        input, enc_w1, enc_b1, h, IN, N0, 0);
    // enc2: bf16 h @ enc_w2 -> bf16 emb, relu
    gemm_wmma<0,true,true><<<dim3(N2 / BN, B / BM), blk, 0, stream>>>(
        h, enc_w2, enc_b2, emb, N0, N2, 0);
    // expert select + matvec + relu: writes f32 latent into d_out, bf16 copy to ws
    expert_kernel<<<dim3(B), blk, 0, stream>>>(emb, td, exp_w, exp_b,
                                               latent_out, latent);
    // dec1: bf16 latent @ dec_w1 -> bf16 d1, relu
    gemm_wmma<0,true,true><<<dim3(N1 / BN, B / BM), blk, 0, stream>>>(
        latent, dec_w1, dec_b1, d1, N2, N1, 0);
    // dec2: bf16 d1 @ dec_w2 -> bf16 d2, relu
    gemm_wmma<0,true,true><<<dim3(N0 / BN, B / BM), blk, 0, stream>>>(
        d1, dec_w2, dec_b2, d2, N1, N0, 0);
    // dec3: bf16 d2 @ dec_w3 -> f32 out with fused means/softplus split
    gemm_wmma<2,true,false><<<dim3((OUTC + BN - 1) / BN, B / BM), blk, 0, stream>>>(
        d2, dec_w3, dec_b3, out, N0, OUTC, IN);
}
